// GNN_EA_56324201120194
// MI455X (gfx1250) — compile-verified, matrix operand order checked by reading, not run on hardware
//
#include <hip/hip_runtime.h>
#include <hip/hip_bf16.h>
#include <math.h>

// Problem constants (from reference)
#define BB   8
#define NN   1024
#define F0S  56
#define DDIM 70
#define DPAD 80    // D padded to multiple of 16 (output/N-dim tiles)
#define KPAD 96    // D padded to multiple of 32 (WMMA K dim)
#define DFCN 128
#define LAYERS 4
#define KCH  256   // K chunk staged in LDS for aggregation GEMM (80*256*2B = 40KB)

typedef _Float16 f16;
typedef _Float16 v16h __attribute__((ext_vector_type(16)));
typedef _Float16 v8h  __attribute__((ext_vector_type(8)));
typedef float    v8f  __attribute__((ext_vector_type(8)));
typedef unsigned int v4u  __attribute__((ext_vector_type(4)));
typedef int          v4i_ __attribute__((ext_vector_type(4)));
typedef int          v8i_ __attribute__((ext_vector_type(8)));

// Load a 16-bit WMMA A/B fragment for this lane from a row-major [rows][ldk]
// f16 array. Per CDNA5 ISA layout (16-bit A 16x32): lanes 0-15 hold row r=lane,
// halves = K{0..7, 16..23}; lanes 16-31 hold row r=lane-16, halves =
// K{8..15, 24..31}. Caller passes p already offset by (row0+r)*ldk + k0 + ko
// where ko = (lane<16 ? 0 : 8). Two contiguous 16-byte loads.
__device__ __forceinline__ v16h load_frag(const f16* p) {
  v8h lo = *(const v8h*)(p);
  v8h hi = *(const v8h*)(p + 16);
  return __builtin_shufflevector(lo, hi, 0,1,2,3,4,5,6,7,8,9,10,11,12,13,14,15);
}

// ---------------------------------------------------------------- embedding
__global__ void k_embed(const float* __restrict__ H, const float* __restrict__ Fw,
                        float* __restrict__ x) {
  int idx = blockIdx.x * blockDim.x + threadIdx.x;
  if (idx >= BB * NN * DDIM) return;
  int d = idx % DDIM;
  int n = idx / DDIM;
  const float* hrow = H + (size_t)n * F0S;
  const float* wrow = Fw + (size_t)d * F0S;
  float s = 0.f;
#pragma unroll
  for (int m = 0; m < F0S; ++m) s += hrow[m] * wrow[m];
  x[idx] = s;
}

// ---------------------------------------------------------------- adjacency
__global__ void k_adj(const float* __restrict__ A1, const float* __restrict__ A2,
                      const float* __restrict__ mu, const float* __restrict__ dev,
                      float* __restrict__ A2r) {
  size_t idx = (size_t)blockIdx.x * blockDim.x + threadIdx.x;
  if (idx >= (size_t)BB * NN * NN) return;
  float a2 = A2[idx];
  float m = mu[0], dv = dev[0];
  float r = (a2 <= 10.f) ? __expf(-(a2 - m) * (a2 - m) / dv) : 0.f;
  A2r[idx] = r + A1[idx];
}

// ---------------------------------------------------------------- h / hA / hT
__global__ void k_h(const float* __restrict__ x, const float* __restrict__ W,
                    const float* __restrict__ A,
                    f16* __restrict__ h16, f16* __restrict__ hA16,
                    f16* __restrict__ hT16) {
  __shared__ float xr[DDIM];
  __shared__ float hr[DDIM];
  int bn = blockIdx.x;              // b*NN + n
  int b = bn >> 10, n = bn & (NN - 1);
  int t = threadIdx.x;
  if (t < DDIM) xr[t] = x[(size_t)bn * DDIM + t];
  __syncthreads();
  float hv = 0.f;
  if (t < DDIM) {
    const float* wr = W + (size_t)t * DDIM;
    for (int m = 0; m < DDIM; ++m) hv += xr[m] * wr[m];
    hr[t] = hv;
  }
  if (t < KPAD) h16[(size_t)bn * KPAD + t] = (t < DDIM) ? (f16)hv : (f16)0.f;
  if (t < DPAD) hT16[((size_t)b * DPAD + t) * NN + n] = (t < DDIM) ? (f16)hv : (f16)0.f;
  __syncthreads();
  if (t < KPAD) {
    float ha = 0.f;
    if (t < DDIM)
      for (int m = 0; m < DDIM; ++m) ha += hr[m] * A[(size_t)m * DDIM + t];
    hA16[(size_t)bn * KPAD + t] = (f16)ha;
  }
}

// ---------------------------------------------------------------- e = hA @ h^T
__global__ void k_e(const f16* __restrict__ hA16, const f16* __restrict__ h16,
                    float* __restrict__ e) {
  int wid = blockIdx.x * (blockDim.x >> 5) + (threadIdx.x >> 5);
  int lane = threadIdx.x & 31;
  int b   = wid >> 10;        // 64 mtiles * 16 ngroups per batch
  int rem = wid & 1023;
  int mt  = rem >> 4;         // 0..63  (M tile)
  int ng  = rem & 15;         // 0..15  (group of 4 N tiles)
  int r  = lane & 15;
  int ko = (lane < 16) ? 0 : 8;
  const f16* Ab = hA16 + ((size_t)b * NN + mt * 16 + r) * KPAD + ko;
  const f16* Bb = h16  + ((size_t)b * NN + ng * 64 + r) * KPAD + ko;
  v8f acc[4] = {};
#pragma unroll
  for (int ks = 0; ks < 3; ++ks) {
    v16h a = load_frag(Ab + ks * 32);
#pragma unroll
    for (int j = 0; j < 4; ++j) {
      v16h bf = load_frag(Bb + (size_t)j * 16 * KPAD + ks * 32);
      acc[j] = __builtin_amdgcn_wmma_f32_16x16x32_f16(false, a, false, bf,
                                                      (short)0, acc[j], false, false);
    }
  }
  int c  = lane & 15;
  int rb = (lane < 16) ? 0 : 8;
  float* eb = e + ((size_t)b * NN + mt * 16 + rb) * NN + ng * 64 + c;
#pragma unroll
  for (int j = 0; j < 4; ++j)
#pragma unroll
    for (int v = 0; v < 8; ++v)
      eb[(size_t)v * NN + j * 16] = acc[j][v];
}

// ---------------------------------------------------------------- symmetrize+mask
__global__ void k_symmask(const float* __restrict__ e, const float* __restrict__ adj,
                          float* __restrict__ S) {
  __shared__ float T1[16][17];
  __shared__ float T2[16][17];
  int tx = threadIdx.x & 15, ty = threadIdx.x >> 4;
  int J = blockIdx.x * 16, K = blockIdx.y * 16, b = blockIdx.z;
  const float* eb = e + (size_t)b * NN * NN;
  T1[ty][tx] = eb[(size_t)(J + ty) * NN + K + tx];
  T2[ty][tx] = eb[(size_t)(K + ty) * NN + J + tx];
  __syncthreads();
  float s = T1[ty][tx] + T2[tx][ty];
  size_t o = (size_t)b * NN * NN + (size_t)(J + ty) * NN + K + tx;
  S[o] = (adj[o] > 0.f) ? s : -9e15f;
}

// ---------------------------------------------------------------- column softmax
__global__ void k_colsoftmax(const float* __restrict__ S,
                             float* __restrict__ Mx, float* __restrict__ Sx) {
  int b = blockIdx.x;
  int c = blockIdx.y * 256 + threadIdx.x;
  const float* Sb = S + (size_t)b * NN * NN + c;
  float m = -__builtin_inff(), s = 0.f;
  for (int j = 0; j < NN; ++j) {
    float v  = Sb[(size_t)j * NN];
    float nm = fmaxf(m, v);
    s = s * __expf(m - nm) + __expf(v - nm);
    m = nm;
  }
  Mx[b * NN + c] = m;
  Sx[b * NN + c] = s;
}

// ---------------------------------------------------------------- probabilities
__global__ void k_prob(const float* __restrict__ S, const float* __restrict__ adj,
                       const float* __restrict__ Mx, const float* __restrict__ Sx,
                       f16* __restrict__ P) {
  size_t idx = (size_t)blockIdx.x * blockDim.x + threadIdx.x;
  if (idx >= (size_t)BB * NN * NN) return;
  int c = (int)(idx & (NN - 1));
  int b = (int)(idx >> 20);
  float att = __expf(S[idx] - Mx[b * NN + c]) / Sx[b * NN + c];
  P[idx] = (f16)(att * adj[idx]);
}

// ---------------------------------------------------------------- aggregation + gate
// hp = relu(P @ h); coeff = sigmoid(x@g1 + hp@g2w + g2b); out = blend.
// B panel (hT, [80][1024] f16) is staged into LDS in K-chunks of 256 by the
// Tensor Data Mover (one TDM descriptor per chunk, issued by wave 0, tracked
// with TENSORcnt); all 8 waves of the block then feed WMMA B fragments from
// LDS. mode 0: xout = val; mode 1: xout -= val.
__global__ void k_agg(const f16* __restrict__ P, const f16* __restrict__ hT,
                      const float* __restrict__ xin, float* __restrict__ xout,
                      const float* __restrict__ g1, const float* __restrict__ g2w,
                      const float* __restrict__ g2b, int mode) {
  __shared__ f16 Bs[DPAD * KCH];          // 40 KB
  int wave = threadIdx.x >> 5;
  int lane = threadIdx.x & 31;
  int b  = blockIdx.x >> 3;               // 8 blocks per batch
  int mt = (blockIdx.x & 7) * 8 + wave;   // M tile 0..63
  int r  = lane & 15;
  int ko = (lane < 16) ? 0 : 8;
  const f16* Ab = P + ((size_t)b * NN + mt * 16 + r) * NN + ko;
  v8f acc[5] = {};
  for (int kc = 0; kc < NN / KCH; ++kc) {
    // ---- stage hT[b][0:80][kc*KCH : (kc+1)*KCH] into Bs (compacted rows) ----
#if __has_builtin(__builtin_amdgcn_tensor_load_to_lds)
    if (wave == 0) {
      unsigned long long ga =
          (unsigned long long)(uintptr_t)(hT + ((size_t)b * DPAD) * NN + kc * KCH);
      unsigned int laddr = (unsigned int)(uintptr_t)(void*)Bs;  // LDS offset (aperture low bits)
      v4u g0;
      g0[0] = 1u;                                  // count = 1 valid descriptor
      g0[1] = laddr;                               // lds_addr (bytes)
      g0[2] = (unsigned int)ga;                    // global_addr[31:0]
      g0[3] = ((unsigned int)(ga >> 32) & 0x01FFFFFFu) | 0x80000000u; // [56:32] | type=2
      v8i_ g1d;
      g1d[0] = (int)(3u << 16);                    // workgroup_mask=0, data_size=8B
      g1d[1] = 0;                                  // abar=0, tensor_dim0[15:0]=0 (2^20)
      g1d[2] = (int)(16u | (80u << 16));           // tensor_dim0[31:16]=16, tensor_dim1=80
      g1d[3] = (int)(64u << 16);                   // tile_dim0 = 64 elems (512B/row)
      g1d[4] = 80;                                 // tile_dim1 = 80 rows, tile_dim2 = 0
      g1d[5] = 256;                                // tensor_dim0_stride = 256 elems (2048B)
      g1d[6] = 0;
      g1d[7] = 0;
      v4i_ gz4 = {0, 0, 0, 0};                     // groups 2/3 unused (2D tensor)
      v8i_ gz8 = {0, 0, 0, 0, 0, 0, 0, 0};         // extra group (clang-23 6-arg form)
      __builtin_amdgcn_tensor_load_to_lds(g0, g1d, gz4, gz4, gz8, 0);
      __builtin_amdgcn_s_wait_tensorcnt(0);
    }
#else
    // Fallback: cooperative vectorized copy (no TDM builtin on this toolchain)
    for (int i = threadIdx.x; i < DPAD * KCH / 8; i += 256) {
      int row = i / (KCH / 8);
      int c8  = i % (KCH / 8);
      *(v8h*)&Bs[row * KCH + c8 * 8] =
          *(const v8h*)(hT + ((size_t)b * DPAD + row) * NN + kc * KCH + c8 * 8);
    }
#endif
    __syncthreads();
    // ---- WMMA over this K chunk: A from global (P), B from LDS ----
    const f16* Abk = Ab + kc * KCH;
    for (int ks = 0; ks < KCH / 32; ++ks) {
      v16h a = load_frag(Abk + ks * 32);
#pragma unroll
      for (int t = 0; t < 5; ++t) {
        v16h bf = load_frag(&Bs[(t * 16 + r) * KCH + ks * 32 + ko]);
        acc[t] = __builtin_amdgcn_wmma_f32_16x16x32_f16(false, a, false, bf,
                                                        (short)0, acc[t], false, false);
      }
    }
    __syncthreads();  // all waves done reading Bs before next chunk overwrite
  }
  // ---- epilogue: relu + gating, all in-register ----
  int c  = lane & 15;
  int rb = (lane < 16) ? 0 : 8;
  float g1v[5], g2v[5];
#pragma unroll
  for (int t = 0; t < 5; ++t) {
    int d = t * 16 + c;
    g1v[t] = (d < DDIM) ? g1[d]  : 0.f;
    g2v[t] = (d < DDIM) ? g2w[d] : 0.f;
  }
  float xv[5][8];
  const float* xb = xin + ((size_t)b * NN + mt * 16 + rb) * DDIM;
#pragma unroll
  for (int t = 0; t < 5; ++t) {
    int d = t * 16 + c;
#pragma unroll
    for (int v = 0; v < 8; ++v)
      xv[t][v] = (d < DDIM) ? xb[(size_t)v * DDIM + d] : 0.f;
  }
  float p1[8], p2[8];
#pragma unroll
  for (int v = 0; v < 8; ++v) {
    float a1 = 0.f, a2 = 0.f;
#pragma unroll
    for (int t = 0; t < 5; ++t) {
      float hp = fmaxf(acc[t][v], 0.f);
      acc[t][v] = hp;
      a1 += xv[t][v] * g1v[t];
      a2 += hp * g2v[t];
    }
    p1[v] = a1; p2[v] = a2;
  }
#pragma unroll
  for (int mask = 1; mask < 16; mask <<= 1) {
#pragma unroll
    for (int v = 0; v < 8; ++v) {
      p1[v] += __shfl_xor(p1[v], mask, 32);
      p2[v] += __shfl_xor(p2[v], mask, 32);
    }
  }
  float bias = g2b[0];
  float* ob = xout + ((size_t)b * NN + mt * 16 + rb) * DDIM;
#pragma unroll
  for (int v = 0; v < 8; ++v) {
    float z = p1[v] + p2[v] + bias;
    float coeff = 1.f / (1.f + __expf(-z));
#pragma unroll
    for (int t = 0; t < 5; ++t) {
      int d = t * 16 + c;
      if (d < DDIM) {
        float val = coeff * xv[t][v] + (1.f - coeff) * acc[t][v];
        float* po = ob + (size_t)v * DDIM + d;
        if (mode == 0) *po = val;
        else           *po = *po - val;
      }
    }
  }
}

// ---------------------------------------------------------------- pooling
__global__ void k_pool(const float* __restrict__ x, const float* __restrict__ V,
                       float* __restrict__ xp) {
  int b = blockIdx.x, t = threadIdx.x;
  if (t < DDIM) {
    float s = 0.f;
    for (int n = 0; n < NN; ++n)
      s += x[((size_t)b * NN + n) * DDIM + t] * V[b * NN + n];
    xp[b * DDIM + t] = s;
  }
}

// ---------------------------------------------------------------- FC head
__global__ void k_fc(const float* __restrict__ xp,
                     const float* __restrict__ w0, const float* __restrict__ b0,
                     const float* __restrict__ w1, const float* __restrict__ b1,
                     const float* __restrict__ w2, const float* __restrict__ b2,
                     const float* __restrict__ w3, const float* __restrict__ b3,
                     float* __restrict__ out) {
  __shared__ float lx[BB * DDIM];
  __shared__ float buf0[BB * DFCN];
  __shared__ float buf1[BB * DFCN];
  int t = threadIdx.x;
  for (int i = t; i < BB * DDIM; i += 256) lx[i] = xp[i];
  __syncthreads();
  for (int i = t; i < BB * DFCN; i += 256) {
    int b = i / DFCN, j = i % DFCN;
    float s = b0[j];
    for (int m = 0; m < DDIM; ++m) s += lx[b * DDIM + m] * w0[j * DDIM + m];
    buf0[i] = fmaxf(s, 0.f);
  }
  __syncthreads();
  for (int i = t; i < BB * DFCN; i += 256) {
    int b = i / DFCN, j = i % DFCN;
    float s = b1[j];
    for (int m = 0; m < DFCN; ++m) s += buf0[b * DFCN + m] * w1[j * DFCN + m];
    buf1[i] = fmaxf(s, 0.f);
  }
  __syncthreads();
  for (int i = t; i < BB * DFCN; i += 256) {
    int b = i / DFCN, j = i % DFCN;
    float s = b2[j];
    for (int m = 0; m < DFCN; ++m) s += buf1[b * DFCN + m] * w2[j * DFCN + m];
    buf0[i] = fmaxf(s, 0.f);
  }
  __syncthreads();
  if (t < BB) {
    float s = b3[0];
    for (int m = 0; m < DFCN; ++m) s += buf0[t * DFCN + m] * w3[m];
    out[t] = 1.f / (1.f + __expf(-s));
  }
}

// ---------------------------------------------------------------- launch
extern "C" void kernel_launch(void* const* d_in, const int* in_sizes, int n_in,
                              void* d_out, int out_size, void* d_ws, size_t ws_size,
                              hipStream_t stream) {
  const float* H    = (const float*)d_in[0];
  const float* A1   = (const float*)d_in[1];
  const float* A2   = (const float*)d_in[2];
  const float* V    = (const float*)d_in[3];
  const float* Fw   = (const float*)d_in[4];
  const float* mu   = (const float*)d_in[5];
  const float* dev  = (const float*)d_in[6];
  const float* gW   = (const float*)d_in[7];
  const float* gA   = (const float*)d_in[8];
  const float* gg1  = (const float*)d_in[9];
  const float* gg2w = (const float*)d_in[10];
  const float* gg2b = (const float*)d_in[11];
  const float* w0 = (const float*)d_in[12]; const float* b0 = (const float*)d_in[13];
  const float* w1 = (const float*)d_in[14]; const float* b1 = (const float*)d_in[15];
  const float* w2 = (const float*)d_in[16]; const float* b2 = (const float*)d_in[17];
  const float* w3 = (const float*)d_in[18]; const float* b3 = (const float*)d_in[19];
  float* outp = (float*)d_out;

  char* ws = (char*)d_ws;
  size_t off = 0;
  auto carve = [&](size_t bytes) -> void* {
    void* p = ws + off;
    off += (bytes + 255) & ~(size_t)255;
    return p;
  };
  float* A2r  = (float*)carve(sizeof(float) * BB * NN * NN);
  float* ebuf = (float*)carve(sizeof(float) * BB * NN * NN);
  float* Sbuf = (float*)carve(sizeof(float) * BB * NN * NN);
  f16*   P16  = (f16*)  carve(sizeof(f16)   * BB * NN * NN);
  f16*   h16  = (f16*)  carve(sizeof(f16)   * BB * NN * KPAD);
  f16*   hA16 = (f16*)  carve(sizeof(f16)   * BB * NN * KPAD);
  f16*   hT16 = (f16*)  carve(sizeof(f16)   * BB * DPAD * NN);
  float* x0   = (float*)carve(sizeof(float) * BB * NN * DDIM);
  float* x1   = (float*)carve(sizeof(float) * BB * NN * DDIM);
  float* Mx   = (float*)carve(sizeof(float) * BB * NN);
  float* Sx   = (float*)carve(sizeof(float) * BB * NN);
  float* xp   = (float*)carve(sizeof(float) * BB * DDIM);

  k_embed<<<(BB * NN * DDIM + 255) / 256, 256, 0, stream>>>(H, Fw, x0);
  k_adj<<<(BB * NN * NN + 255) / 256, 256, 0, stream>>>(A1, A2, mu, dev, A2r);

  float* xcur = x0;
  float* xnxt = x1;
  for (int k = 0; k < LAYERS; ++k) {
    const float* Wk   = gW   + k * DDIM * DDIM;
    const float* Ak   = gA   + k * DDIM * DDIM;
    const float* g1k  = gg1  + k * DDIM;
    const float* g2wk = gg2w + k * DDIM;
    const float* g2bk = gg2b + k;
    const float* adjs[2] = { A2r, A1 };
    for (int m = 0; m < 2; ++m) {
      k_h<<<BB * NN, 128, 0, stream>>>(xcur, Wk, Ak, h16, hA16, hT16);
      k_e<<<1024, 256, 0, stream>>>(hA16, h16, ebuf);             // 8192 waves
      k_symmask<<<dim3(64, 64, BB), 256, 0, stream>>>(ebuf, adjs[m], Sbuf);
      k_colsoftmax<<<dim3(BB, 4), 256, 0, stream>>>(Sbuf, Mx, Sx);
      k_prob<<<(BB * NN * NN + 255) / 256, 256, 0, stream>>>(Sbuf, adjs[m], Mx, Sx, P16);
      k_agg<<<64, 256, 0, stream>>>(P16, hT16, xcur, xnxt, g1k, g2wk, g2bk, m); // 512 waves
    }
    float* t = xcur; xcur = xnxt; xnxt = t;
  }
  k_pool<<<BB, 128, 0, stream>>>(xcur, V, xp);
  k_fc<<<1, 256, 0, stream>>>(xp, w0, b0, w1, b1, w2, b2, w3, b3, outp);
}
